// Point_Transformer_Last_27350351741697
// MI455X (gfx1250) — matrix operand order repaired and checked by reading, not compiled, not run on hardware
//
#include <hip/hip_runtime.h>
#include <hip/hip_bf16.h>

#define DEVI __device__ __forceinline__

typedef __bf16 bf16_t;
typedef __attribute__((ext_vector_type(16))) __bf16 v16bf;
typedef __attribute__((ext_vector_type(8))) float v8f;

union FragAB {
  v16bf v;
  unsigned int u[8];
  unsigned short s[16];
};
union FragC {
  v8f v;
  float f[8];
};

constexpr int CC = 256;   // channels
constexpr int DD = 64;    // shared q/k dim
constexpr int NN = 4096;  // points
constexpr int BB = 4;     // batch
constexpr float EPS_BN = 1e-5f;

DEVI unsigned short f2bf(float x) {
  unsigned int u = __float_as_uint(x);
  u += 0x7FFFu + ((u >> 16) & 1u);
  return (unsigned short)(u >> 16);
}

DEVI v8f wmma_bf16(const FragAB& a, const FragAB& b, v8f c) {
  return __builtin_amdgcn_wmma_f32_16x16x32_bf16(false, a.v, false, b.v, (short)0, c,
                                                 false, false);
}

// CDNA5 async global->LDS copy, 16 bytes per lane (ASYNCcnt tracked).
// dst: generic pointer to __shared__ (low 32 bits = LDS byte address).
DEVI void async_copy_b128(unsigned short* lds_dst, const unsigned short* gsrc) {
  unsigned lds_addr = (unsigned)(unsigned long long)lds_dst;
  asm volatile("global_load_async_to_lds_b128 %0, %1, off"
               :: "v"(lds_addr), "v"(gsrc)
               : "memory");
}
DEVI void wait_async0() { asm volatile("s_wait_asynccnt 0" ::: "memory"); }

// A-operand frag (16x32 bf16) from LDS tile stored [row][k] (k contiguous).
DEVI FragAB lds_load_a(const unsigned short* p0, int rowBase, int strideUS, int kOff) {
  const int lane = threadIdx.x & 31;
  const unsigned short* p =
      p0 + (rowBase + (lane & 15)) * strideUS + kOff + ((lane & 16) ? 8 : 0);
  FragAB f;
#pragma unroll
  for (int j = 0; j < 4; ++j) {
    f.u[j]     = *(const unsigned int*)(p + 2 * j);
    f.u[4 + j] = *(const unsigned int*)(p + 16 + 2 * j);
  }
  return f;
}

// B-operand frag (32x16 bf16) from LDS tile stored transposed [n][k].
DEVI FragAB lds_load_b(const unsigned short* p0, int colBase, int strideUS, int kOff) {
  const int lane = threadIdx.x & 31;
  const unsigned short* p =
      p0 + (colBase + (lane & 15)) * strideUS + kOff + ((lane & 16) ? 8 : 0);
  FragAB f;
#pragma unroll
  for (int j = 0; j < 4; ++j) {
    f.u[j]     = *(const unsigned int*)(p + 2 * j);
    f.u[4 + j] = *(const unsigned int*)(p + 16 + 2 * j);
  }
  return f;
}

// A-operand frag directly from global row-major [row][k], row stride rs.
DEVI FragAB glb_load_a(const unsigned short* p0, int rowBase, size_t rs, int kBase) {
  const int lane = threadIdx.x & 31;
  const unsigned short* p =
      p0 + (size_t)(rowBase + (lane & 15)) * rs + kBase + ((lane & 16) ? 8 : 0);
  FragAB f;
#pragma unroll
  for (int j = 0; j < 4; ++j) {
    f.u[j]     = *(const unsigned int*)(p + 2 * j);
    f.u[4 + j] = *(const unsigned int*)(p + 16 + 2 * j);
  }
  return f;
}

// ---------------------------------------------------------------------------
// Generic 256-K WMMA GEMM: Out[M x NN] = W[M x 256] * X[256 x NN]  (per batch z)
// MODE 0: BN+ReLU -> outF; MODE 1: +bias -> outB(bf16);
// MODE 2: +bias,BN,ReLU,+resid -> outF AND out2; MODE 3: -> outB + outBT (bf16)
// ---------------------------------------------------------------------------
template <int BM, int BN, int MODE>
__global__ __launch_bounds__(256) void gemm_kernel(
    const unsigned short* __restrict__ W, const unsigned short* __restrict__ X,
    const float* __restrict__ bias, const float* __restrict__ bng,
    const float* __restrict__ bnb, const float* __restrict__ bnm,
    const float* __restrict__ bnv, const float* __restrict__ resid,
    float* __restrict__ outF, unsigned short* __restrict__ outB,
    unsigned short* __restrict__ outBT, float* __restrict__ out2, int layerC) {
  constexpr int K = 256;
  constexpr int AST = 40;                  // A row stride (ushorts), 80B: 16B-aligned
  constexpr int BST = 40;                  // BT row stride
  constexpr int MT = BM / 16;
  constexpr int NTW = (BN / 16) * MT / 8;  // col tiles per wave

  __shared__ unsigned short lA[BM * AST];
  __shared__ unsigned short lBT[BN * BST];

  const int tid = threadIdx.x;
  const int wid = tid >> 5;
  const int lane = tid & 31;
  const int b = blockIdx.z;
  const int mBase = blockIdx.y * BM;
  const int nBase = blockIdx.x * BN;
  const unsigned short* Xb = X + (size_t)b * K * NN;

  const int mt = wid % MT;
  const int ctb = (wid / MT) * NTW;

  FragC acc[NTW];
#pragma unroll
  for (int j = 0; j < NTW; ++j)
#pragma unroll
    for (int r = 0; r < 8; ++r) acc[j].f[r] = 0.f;

  for (int k0 = 0; k0 < K; k0 += 32) {
    __syncthreads();
    // A tile: BM x 32, straight copy -> async to LDS
#pragma unroll
    for (int l = 0; l < BM * 4 / 256; ++l) {
      int idx = tid + l * 256;
      int row = idx >> 2, seg = idx & 3;
      async_copy_b128(&lA[row * AST + seg * 8],
                      W + (size_t)(mBase + row) * K + k0 + seg * 8);
    }
    // B tile: 32 x BN, stored transposed [n][k]
    {
      constexpr int TPN = 256 / BN;
      constexpr int KPT = 32 / TPN;
      int n = tid % BN;
      int kk = (tid / BN) * KPT;
      unsigned short tb[KPT];
#pragma unroll
      for (int j = 0; j < KPT; ++j)
        tb[j] = Xb[(size_t)(k0 + kk + j) * NN + nBase + n];
      if (k0 + 32 < K) __builtin_prefetch(Xb + (size_t)(k0 + 32 + kk) * NN + nBase + n);
#pragma unroll
      for (int j = 0; j < KPT; j += 4) {
        uint2 pk;
        pk.x = (unsigned int)tb[j] | ((unsigned int)tb[j + 1] << 16);
        pk.y = (unsigned int)tb[j + 2] | ((unsigned int)tb[j + 3] << 16);
        *(uint2*)&lBT[n * BST + kk + j] = pk;
      }
    }
    wait_async0();
    __syncthreads();
    FragAB a = lds_load_a(lA, mt * 16, AST, 0);
#pragma unroll
    for (int j = 0; j < NTW; ++j) {
      FragAB bf = lds_load_b(lBT, (ctb + j) * 16, BST, 0);
      acc[j].v = wmma_bf16(a, bf, acc[j].v);
    }
  }

  const int nloc = lane & 15;
  const int half8 = (lane & 16) ? 8 : 0;
#pragma unroll
  for (int j = 0; j < NTW; ++j) {
    const int ng = nBase + (ctb + j) * 16 + nloc;
#pragma unroll
    for (int r = 0; r < 8; ++r) {
      const int mg = mBase + mt * 16 + half8 + r;
      float val = acc[j].f[r];
      if (MODE == 0) {
        float inv = bng[mg] * rsqrtf(bnv[mg] + EPS_BN);
        val = fmaxf(val * inv + (bnb[mg] - bnm[mg] * inv), 0.f);
        outF[(size_t)b * CC * NN + (size_t)mg * NN + ng] = val;
      } else if (MODE == 1) {
        val += bias[mg];
        outB[(size_t)b * CC * NN + (size_t)mg * NN + ng] = f2bf(val);
      } else if (MODE == 2) {
        val += bias[mg];
        float inv = bng[mg] * rsqrtf(bnv[mg] + EPS_BN);
        val = fmaxf(val * inv + (bnb[mg] - bnm[mg] * inv), 0.f);
        val += resid[(size_t)b * CC * NN + (size_t)mg * NN + ng];
        outF[(size_t)b * CC * NN + (size_t)mg * NN + ng] = val;
        out2[((size_t)b * (4 * CC) + layerC + mg) * NN + ng] = val;
      } else {  // MODE 3
        unsigned short hv = f2bf(val);
        outB[(size_t)b * DD * NN + (size_t)mg * NN + ng] = hv;
        outBT[((size_t)b * NN + ng) * DD + mg] = hv;
      }
    }
  }
}

// ---------------------------------------------------------------------------
// Pass A: softmax stats via two sweeps over recomputed Gram energy.
// Sweep 1: per-lane running max (no shuffles in loop). Reduce once.
// Sweep 2: per-lane sum of exp(e - rowmax). Reduce once.
// ---------------------------------------------------------------------------
__global__ __launch_bounds__(256) void attn_pass_a(
    const unsigned short* __restrict__ qk,   // [B][64][NN]
    const unsigned short* __restrict__ qkT,  // [B][NN][64]
    float* __restrict__ rowMax, float* __restrict__ rowSum) {
  constexpr int QST = 72;  // 144B rows: 16B-aligned for async b128
  __shared__ unsigned short lA[64 * QST];
  __shared__ unsigned short lB[128 * QST];
  __shared__ float sStat[2][64];
  __shared__ float sMax[64];

  const int tid = threadIdx.x, wid = tid >> 5, lane = tid & 31;
  const int b = blockIdx.z;
  const int rowBase = blockIdx.x * 64;

  // A tile (qkT rows): straight copy -> async
#pragma unroll
  for (int l = 0; l < 2; ++l) {
    int idx = tid + l * 256;
    int row = idx >> 3, seg = idx & 7;
    async_copy_b128(&lA[row * QST + seg * 8],
                    qkT + ((size_t)b * NN + rowBase + row) * DD + seg * 8);
  }
  wait_async0();
  __syncthreads();

  const int rt = wid & 3;
  const int cg = wid >> 2;
  const int half8 = (lane & 16) ? 8 : 0;
  FragAB a0 = lds_load_a(lA, rt * 16, QST, 0);
  FragAB a1 = lds_load_a(lA, rt * 16, QST, 32);

  auto stageB = [&](int m0) {
#pragma unroll
    for (int l = 0; l < 4; ++l) {
      int m = tid & 127;
      int d0 = ((tid >> 7) * 8) + l * 16;
      unsigned short tb[8];
#pragma unroll
      for (int j = 0; j < 8; ++j)
        tb[j] = qk[((size_t)b * DD + d0 + j) * NN + m0 + m];
      uint2* dst = (uint2*)&lB[m * QST + d0];
      dst[0] = make_uint2((unsigned)tb[0] | ((unsigned)tb[1] << 16),
                          (unsigned)tb[2] | ((unsigned)tb[3] << 16));
      dst[1] = make_uint2((unsigned)tb[4] | ((unsigned)tb[5] << 16),
                          (unsigned)tb[6] | ((unsigned)tb[7] << 16));
    }
  };

  // ---- sweep 1: row max ----
  float rmax[8];
#pragma unroll
  for (int r = 0; r < 8; ++r) rmax[r] = -3.0e38f;
  for (int m0 = 0; m0 < NN; m0 += 128) {
    __syncthreads();
    stageB(m0);
    __syncthreads();
#pragma unroll
    for (int j = 0; j < 4; ++j) {
      int mtile = cg * 4 + j;
      FragC e;
#pragma unroll
      for (int r = 0; r < 8; ++r) e.f[r] = 0.f;
      e.v = wmma_bf16(a0, lds_load_b(lB, mtile * 16, QST, 0), e.v);
      e.v = wmma_bf16(a1, lds_load_b(lB, mtile * 16, QST, 32), e.v);
#pragma unroll
      for (int r = 0; r < 8; ++r) rmax[r] = fmaxf(rmax[r], e.f[r]);
    }
  }
#pragma unroll
  for (int r = 0; r < 8; ++r)
#pragma unroll
    for (int msk = 1; msk <= 8; msk <<= 1) rmax[r] = fmaxf(rmax[r], __shfl_xor(rmax[r], msk));
  if (lane == 0 || lane == 16) {
#pragma unroll
    for (int r = 0; r < 8; ++r) sStat[cg][rt * 16 + half8 + r] = rmax[r];
  }
  __syncthreads();
  if (tid < 64) sMax[tid] = fmaxf(sStat[0][tid], sStat[1][tid]);
  __syncthreads();
  float rmx[8];
#pragma unroll
  for (int r = 0; r < 8; ++r) rmx[r] = sMax[rt * 16 + half8 + r];

  // ---- sweep 2: row sum of exp ----
  float rsum[8];
#pragma unroll
  for (int r = 0; r < 8; ++r) rsum[r] = 0.f;
  for (int m0 = 0; m0 < NN; m0 += 128) {
    __syncthreads();
    stageB(m0);
    __syncthreads();
#pragma unroll
    for (int j = 0; j < 4; ++j) {
      int mtile = cg * 4 + j;
      FragC e;
#pragma unroll
      for (int r = 0; r < 8; ++r) e.f[r] = 0.f;
      e.v = wmma_bf16(a0, lds_load_b(lB, mtile * 16, QST, 0), e.v);
      e.v = wmma_bf16(a1, lds_load_b(lB, mtile * 16, QST, 32), e.v);
#pragma unroll
      for (int r = 0; r < 8; ++r) rsum[r] += __expf(e.f[r] - rmx[r]);
    }
  }
#pragma unroll
  for (int r = 0; r < 8; ++r)
#pragma unroll
    for (int msk = 1; msk <= 8; msk <<= 1) rsum[r] += __shfl_xor(rsum[r], msk);
  if (lane == 0 || lane == 16) {
#pragma unroll
    for (int r = 0; r < 8; ++r) sStat[cg][rt * 16 + half8 + r] = rsum[r];
  }
  __syncthreads();
  if (tid < 64) {
    rowMax[(size_t)b * NN + rowBase + tid] = sMax[tid];
    rowSum[(size_t)b * NN + rowBase + tid] = sStat[0][tid] + sStat[1][tid];
  }
}

// ---------------------------------------------------------------------------
// Pass B: per 64-col chunk, stream n-blocks: energy -> att tile (bf16 LDS),
// y += V*att (WMMA), colsum accumulated in registers. Epilogue:
// d = h - y/(1e-9+colsum) -> dbf (bf16).
// ---------------------------------------------------------------------------
__global__ __launch_bounds__(256) void attn_pass_b(
    const unsigned short* __restrict__ qk, const unsigned short* __restrict__ qkT,
    const unsigned short* __restrict__ vbf, const float* __restrict__ rowMax,
    const float* __restrict__ rowSum, const float* __restrict__ h,
    unsigned short* __restrict__ dbf) {
  constexpr int QST = 72;
  __shared__ unsigned short lBm[64 * QST];   // qk m-chunk, [m][d]
  __shared__ unsigned short lAq[64 * QST];   // qkT n-block, [n][d]
  __shared__ unsigned short lAtt[64 * QST];  // att, [m][n]
  __shared__ float sCol[64];
  __shared__ float sRM[64], sRSi[64];

  const int tid = threadIdx.x, wid = tid >> 5, lane = tid & 31;
  const int b = blockIdx.z;
  const int mBase = blockIdx.x * 64;

  // stage qk m-chunk transposed -> lBm[m][d] (once)
#pragma unroll
  for (int l = 0; l < 2; ++l) {
    int m = tid & 63;
    int d0 = ((tid >> 6) * 8) + l * 32;
    unsigned short tb[8];
#pragma unroll
    for (int j = 0; j < 8; ++j)
      tb[j] = qk[((size_t)b * DD + d0 + j) * NN + mBase + m];
    uint2* dst = (uint2*)&lBm[m * QST + d0];
    dst[0] = make_uint2((unsigned)tb[0] | ((unsigned)tb[1] << 16),
                        (unsigned)tb[2] | ((unsigned)tb[3] << 16));
    dst[1] = make_uint2((unsigned)tb[4] | ((unsigned)tb[5] << 16),
                        (unsigned)tb[6] | ((unsigned)tb[7] << 16));
  }
  if (tid < 64) sCol[tid] = 0.f;

  FragC acc[2][4];
#pragma unroll
  for (int ci = 0; ci < 2; ++ci)
#pragma unroll
    for (int mj = 0; mj < 4; ++mj)
#pragma unroll
      for (int r = 0; r < 8; ++r) acc[ci][mj].f[r] = 0.f;

  const int ntE = wid & 3;
  const int mtE0 = (wid >> 2) * 2;
  const int cBase = wid * 32;
  float csum[2] = {0.f, 0.f};

  for (int nb = 0; nb < NN; nb += 64) {
    __syncthreads();
#pragma unroll
    for (int l = 0; l < 2; ++l) {
      int idx = tid + l * 256;
      int row = idx >> 3, seg = idx & 7;
      async_copy_b128(&lAq[row * QST + seg * 8],
                      qkT + ((size_t)b * NN + nb + row) * DD + seg * 8);
    }
    if (tid < 64) {
      sRM[tid] = rowMax[(size_t)b * NN + nb + tid];
      sRSi[tid] = 1.0f / rowSum[(size_t)b * NN + nb + tid];
    }
    wait_async0();
    __syncthreads();
    // phase 1: energy -> att tile
    {
      FragAB a0 = lds_load_a(lAq, ntE * 16, QST, 0);
      FragAB a1 = lds_load_a(lAq, ntE * 16, QST, 32);
#pragma unroll
      for (int t = 0; t < 2; ++t) {
        int mt = mtE0 + t;
        FragC e;
#pragma unroll
        for (int r = 0; r < 8; ++r) e.f[r] = 0.f;
        e.v = wmma_bf16(a0, lds_load_b(lBm, mt * 16, QST, 0), e.v);
        e.v = wmma_bf16(a1, lds_load_b(lBm, mt * 16, QST, 32), e.v);
        int mcol = mt * 16 + (lane & 15);
#pragma unroll
        for (int r = 0; r < 8; ++r) {
          int nrow = ntE * 16 + ((lane & 16) ? 8 : 0) + r;
          float av = __expf(e.f[r] - sRM[nrow]) * sRSi[nrow];
          csum[t] += av;
          lAtt[mcol * QST + nrow] = f2bf(av);
        }
      }
    }
    __syncthreads();
    // phase 2: y += V * att
#pragma unroll
    for (int ks = 0; ks < 2; ++ks) {
      FragAB bfr[4];
#pragma unroll
      for (int mj = 0; mj < 4; ++mj) bfr[mj] = lds_load_b(lAtt, mj * 16, QST, ks * 32);
#pragma unroll
      for (int ci = 0; ci < 2; ++ci) {
        FragAB av = glb_load_a(vbf + (size_t)b * CC * NN, cBase + ci * 16, NN, nb + ks * 32);
#pragma unroll
        for (int mj = 0; mj < 4; ++mj)
          acc[ci][mj].v = wmma_bf16(av, bfr[mj], acc[ci][mj].v);
      }
    }
  }
  // merge colsum partials (per lane, once)
#pragma unroll
  for (int t = 0; t < 2; ++t) {
    csum[t] += __shfl_xor(csum[t], 16);
    if (lane < 16) atomicAdd(&sCol[(mtE0 + t) * 16 + (lane & 15)], csum[t]);
  }
  __syncthreads();
#pragma unroll
  for (int ci = 0; ci < 2; ++ci) {
#pragma unroll
    for (int mj = 0; mj < 4; ++mj) {
      int ng = mBase + mj * 16 + (lane & 15);
      float rinv = 1.0f / (1e-9f + sCol[mj * 16 + (lane & 15)]);
#pragma unroll
      for (int r = 0; r < 8; ++r) {
        int cgr = cBase + ci * 16 + ((lane & 16) ? 8 : 0) + r;
        float xr = acc[ci][mj].f[r] * rinv;
        float dv = h[(size_t)b * CC * NN + (size_t)cgr * NN + ng] - xr;
        dbf[(size_t)b * CC * NN + (size_t)cgr * NN + ng] = f2bf(dv);
      }
    }
  }
}

// ---------------------------------------------------------------------------
// Small elementwise kernels
// ---------------------------------------------------------------------------
__global__ void cvt_bf16_kernel(const float* __restrict__ src,
                                unsigned short* __restrict__ dst, int count) {
  int i = blockIdx.x * 256 + threadIdx.x;
  if (i < count) dst[i] = f2bf(src[i]);
}

__global__ void pos_kernel(const float* __restrict__ xyz, const float* __restrict__ pw,
                           const float* __restrict__ pb, float* __restrict__ pos) {
  int i = blockIdx.x * 256 + threadIdx.x;  // B*C*N total
  int n = i & (NN - 1);
  int c = (i >> 12) & (CC - 1);
  int b = i >> 20;
  const float* q = xyz + ((size_t)b * NN + n) * 3;
  pos[i] = pw[c * 3] * q[0] + pw[c * 3 + 1] * q[1] + pw[c * 3 + 2] * q[2] + pb[c];
}

__global__ void hpx_kernel(const float* __restrict__ h, const float* __restrict__ pos,
                           unsigned short* __restrict__ out) {
  int i = blockIdx.x * 256 + threadIdx.x;
  out[i] = f2bf(h[i] + pos[i]);
}

// ---------------------------------------------------------------------------
extern "C" void kernel_launch(void* const* d_in, const int* in_sizes, int n_in,
                              void* d_out, int out_size, void* d_ws, size_t ws_size,
                              hipStream_t stream) {
  const float* x       = (const float*)d_in[0];
  const float* xyz     = (const float*)d_in[1];
  const float* pos_w   = (const float*)d_in[2];
  const float* pos_b   = (const float*)d_in[3];
  const float* conv1_w = (const float*)d_in[4];
  const float* bn1_g   = (const float*)d_in[5];
  const float* bn1_b   = (const float*)d_in[6];
  const float* bn1_m   = (const float*)d_in[7];
  const float* bn1_v   = (const float*)d_in[8];
  const float* qk_w    = (const float*)d_in[9];
  const float* v_w     = (const float*)d_in[10];
  const float* v_b     = (const float*)d_in[11];
  const float* tr_w    = (const float*)d_in[12];
  const float* tr_b    = (const float*)d_in[13];
  const float* bn_g    = (const float*)d_in[14];
  const float* bn_b    = (const float*)d_in[15];
  const float* bn_m    = (const float*)d_in[16];
  const float* bn_v    = (const float*)d_in[17];
  float* out = (float*)d_out;

  char* ws = (char*)d_ws;
  size_t off = 0;
  auto alloc = [&](size_t bytes) -> char* {
    char* p = ws + off;
    off += (bytes + 255) & ~(size_t)255;
    return p;
  };

  const size_t tot = (size_t)BB * CC * NN;  // 4M
  float* pos            = (float*)alloc(tot * 4);
  float* h              = (float*)alloc(tot * 4);
  unsigned short* hpx   = (unsigned short*)alloc(tot * 2);  // also holds bf16(x)
  unsigned short* vbf   = (unsigned short*)alloc(tot * 2);
  unsigned short* dbf   = (unsigned short*)alloc(tot * 2);
  unsigned short* qkb   = (unsigned short*)alloc((size_t)BB * DD * NN * 2);
  unsigned short* qkT   = (unsigned short*)alloc((size_t)BB * NN * DD * 2);
  float* rowMax         = (float*)alloc((size_t)BB * NN * 4);
  float* rowSum         = (float*)alloc((size_t)BB * NN * 4);
  unsigned short* wConv = (unsigned short*)alloc(65536 * 2);
  unsigned short* wQk   = (unsigned short*)alloc(65536 * 2);
  unsigned short* wV    = (unsigned short*)alloc(262144 * 2);
  unsigned short* wTr   = (unsigned short*)alloc(262144 * 2);

  cvt_bf16_kernel<<<(int)(tot / 256), 256, 0, stream>>>(x, hpx, (int)tot);
  cvt_bf16_kernel<<<65536 / 256, 256, 0, stream>>>(conv1_w, wConv, 65536);
  cvt_bf16_kernel<<<65536 / 256, 256, 0, stream>>>(qk_w, wQk, 65536);
  cvt_bf16_kernel<<<262144 / 256, 256, 0, stream>>>(v_w, wV, 262144);
  cvt_bf16_kernel<<<262144 / 256, 256, 0, stream>>>(tr_w, wTr, 262144);
  pos_kernel<<<(int)(tot / 256), 256, 0, stream>>>(xyz, pos_w, pos_b, pos);

  dim3 g256(NN / 64, 2, BB);
  gemm_kernel<128, 64, 0><<<g256, 256, 0, stream>>>(
      wConv, hpx, nullptr, bn1_g, bn1_b, bn1_m, bn1_v, nullptr, h, nullptr, nullptr,
      nullptr, 0);

  for (int i = 0; i < 4; ++i) {
    hpx_kernel<<<(int)(tot / 256), 256, 0, stream>>>(h, pos, hpx);
    dim3 gqk(NN / 128, 1, BB);
    gemm_kernel<64, 128, 3><<<gqk, 256, 0, stream>>>(
        wQk + (size_t)i * DD * CC, hpx, nullptr, nullptr, nullptr, nullptr, nullptr,
        nullptr, nullptr, qkb, qkT, nullptr, 0);
    gemm_kernel<128, 64, 1><<<g256, 256, 0, stream>>>(
        wV + (size_t)i * CC * CC, hpx, v_b + i * CC, nullptr, nullptr, nullptr, nullptr,
        nullptr, nullptr, vbf, nullptr, nullptr, 0);
    attn_pass_a<<<dim3(NN / 64, 1, BB), 256, 0, stream>>>(qkb, qkT, rowMax, rowSum);
    attn_pass_b<<<dim3(NN / 64, 1, BB), 256, 0, stream>>>(qkb, qkT, vbf, rowMax, rowSum,
                                                          h, dbf);
    gemm_kernel<128, 64, 2><<<g256, 256, 0, stream>>>(
        wTr + (size_t)i * CC * CC, dbf, tr_b + i * CC, bn_g + i * CC, bn_b + i * CC,
        bn_m + i * CC, bn_v + i * CC, h, h, nullptr, nullptr, out, i * CC);
  }
}
